// ProjectedConjugatedCSPNet_65034394796451
// MI455X (gfx1250) — matrix operand order repaired and checked by reading, not compiled, not run on hardware
//
#include <hip/hip_runtime.h>

// ---------------------------------------------------------------------------
// Problem constants (match reference)
// ---------------------------------------------------------------------------
#define NN   10000      // nodes
#define NE   100000     // edges
#define HD   512        // hidden
#define TDD  256        // time-emb dim
#define DINF 1798       // edge-mlp input dim (512 src + 512 dst + 6 lat + 768 demb)
#define DISF 768        // sinusoid dim
#define NL   2          // layers
#define KE_CH 56        // padded edge K (1792 = 512+512+768) / 32
#define KN_CH 32        // node K (1024) / 32
#define KH_CH 16        // hidden K (512) / 32
#define EM   64         // edges per workgroup (4 m-tiles): halves weight re-reads

typedef __attribute__((ext_vector_type(16))) _Float16 v16h;
typedef __attribute__((ext_vector_type(8)))  float    v8f;

union AFrag { v16h h; unsigned int u[8]; };

__device__ __forceinline__ float siluf(float x) { return x / (1.0f + __expf(-x)); }

__device__ __forceinline__ float waveRed(float v) {
#pragma unroll
  for (int i = 16; i > 0; i >>= 1) v += __shfl_xor(v, i, 32);
  return v;
}

__device__ __forceinline__ void atomAddF(float* p, float v) {
  // lowers to global_atomic_add_f32 on gfx1250
  __hip_atomic_fetch_add(p, v, __ATOMIC_RELAXED, __HIP_MEMORY_SCOPE_AGENT);
}

// ---------------------------------------------------------------------------
// Generic zero
// ---------------------------------------------------------------------------
__global__ __launch_bounds__(256) void zero_kernel(float* p, int n) {
  int i = blockIdx.x * 256 + threadIdx.x;
  if (i < n) p[i] = 0.0f;
}

// ---------------------------------------------------------------------------
// Pack a [L, 512, Kfull] f32 weight (row-major, out-features x in-features)
// into per-tile WMMA B fragments (f16).  B tile = 32(K) x 16(N).
// Fragment layout (ISA 7.12.2, B mirrors A with N across lanes):
//   lane l: n = l%16 ; k_within = (j<8 ? 0 : 16) + (l>=16 ? 8 : 0) + (j&7)   (j = 0..15 halves)
// dst element index = ((layer*32 + nt)*nk + kc)*512 + lane*16 + j
// remap==1: skip the 6 lattice columns of eW1 (packed k>=1024 -> col k+6)
// ---------------------------------------------------------------------------
__global__ __launch_bounds__(256) void pack_w_kernel(const float* __restrict__ W,
                                                     _Float16* __restrict__ dst,
                                                     int nk, int Kfull, int remap,
                                                     int total) {
  int idx = blockIdx.x * 256 + threadIdx.x;
  if (idx >= total) return;
  int j    = idx & 15;
  int lane = (idx >> 4) & 31;
  int tile = idx >> 9;
  int tilesPerLayer = 32 * nk;
  int layer = tile / tilesPerLayer;
  int tl    = tile % tilesPerLayer;
  int nt = tl / nk, kc = tl % nk;
  int n = nt * 16 + (lane & 15);
  int k = kc * 32 + ((j < 8) ? 0 : 16) + ((lane >= 16) ? 8 : 0) + (j & 7);
  int col = (remap && k >= 1024) ? (k + 6) : k;
  dst[idx] = (_Float16)W[((size_t)layer * HD + n) * (size_t)Kfull + col];
}

// ---------------------------------------------------------------------------
// Sinusoid embedding: demb[e, j] = sin(x_d * 2*pi*f) for j<384, cos for j>=384.
// One thread per angle: sincos once, two f16 stores.
// ---------------------------------------------------------------------------
__global__ __launch_bounds__(256) void demb_kernel(const float* __restrict__ fd,
                                                   _Float16* __restrict__ demb16) {
  long long idx = (long long)blockIdx.x * 256 + threadIdx.x;
  if (idx >= (long long)NE * 384) return;
  int e  = (int)(idx / 384);
  int jj = (int)(idx % 384);
  int d = jj >> 7, f = jj & 127;
  float ang = fd[e * 3 + d] * (6.28318530717958647692f * (float)f);
  float s, c;
  __sincosf(ang, &s, &c);
  demb16[(size_t)e * DISF + jj]       = (_Float16)s;
  demb16[(size_t)e * DISF + 384 + jj] = (_Float16)c;
}

// ---------------------------------------------------------------------------
// counts[src] += 1 per edge (scatter-mean denominator)
// ---------------------------------------------------------------------------
__global__ __launch_bounds__(256) void counts_kernel(const int* __restrict__ eidx,
                                                     float* __restrict__ counts) {
  int i = blockIdx.x * 256 + threadIdx.x;
  if (i < NE) atomAddF(&counts[eidx[i]], 1.0f);
}

// ---------------------------------------------------------------------------
// Node init: h = silu([emb_table[atom], t*Wt] @ Wa.T + ba)   (8 GFLOP, VALU ok)
// ---------------------------------------------------------------------------
__global__ __launch_bounds__(256) void init_h_kernel(const int* __restrict__ atom,
                                                     const float* __restrict__ emb,
                                                     const float* __restrict__ Wt,
                                                     const float* __restrict__ Wa,
                                                     const float* __restrict__ ba,
                                                     const float* __restrict__ t,
                                                     float* __restrict__ h) {
  int idx = blockIdx.x * 256 + threadIdx.x;
  if (idx >= NN * HD) return;
  int node = idx >> 9, c = idx & 511;
  const float* wrow = Wa + (size_t)c * (HD + TDD);
  const float* erow = emb + (size_t)atom[node] * HD;
  float t0 = t[0];
  float s = ba[c];
  for (int k = 0; k < HD; ++k)  s += erow[k] * wrow[k];
  for (int k = 0; k < TDD; ++k) s += (t0 * Wt[k]) * wrow[HD + k];
  h[idx] = siluf(s);
}

// ---------------------------------------------------------------------------
// LayerNorm -> f16 (one wave32 per node; 16 elements per lane)
// ---------------------------------------------------------------------------
__global__ __launch_bounds__(256) void ln_kernel(const float* __restrict__ h,
                                                 const float* __restrict__ g,
                                                 const float* __restrict__ b,
                                                 _Float16* __restrict__ out) {
  int wave = threadIdx.x >> 5, lane = threadIdx.x & 31;
  int node = blockIdx.x * 8 + wave;
  if (node >= NN) return;
  const float* row = h + (size_t)node * HD;
  float x[16], s = 0.f, s2 = 0.f;
#pragma unroll
  for (int i = 0; i < 16; ++i) {
    float v = row[lane + i * 32];
    x[i] = v; s += v; s2 += v * v;
  }
  s = waveRed(s); s2 = waveRed(s2);
  float mu = s * (1.0f / HD);
  float var = s2 * (1.0f / HD) - mu * mu;
  float rs = rsqrtf(var + 1e-5f);
#pragma unroll
  for (int i = 0; i < 16; ++i) {
    int c = lane + i * 32;
    out[(size_t)node * HD + c] = (_Float16)((x[i] - mu) * rs * g[c] + b[c]);
  }
}

// ---------------------------------------------------------------------------
// One K-chunk of the edge GEMM1: stage 64x32 A slab (global f16 -> LDS via one
// uint4 per thread), build A fragments, 16 WMMAs against pre-packed B tiles.
// ---------------------------------------------------------------------------
__device__ __forceinline__ void edge_chunk(const _Float16* __restrict__ stageSrc,
                                           _Float16* __restrict__ stageDst,
                                           const _Float16* __restrict__ aT,
                                           const _Float16* __restrict__ wp,
                                           int nk, int kc, int lane, int wave,
                                           v8f (&acc)[4][4]) {
  __syncthreads();                                  // WAR on aT
  *(uint4*)stageDst = *(const uint4*)stageSrc;      // 16B per thread
  __syncthreads();
  AFrag a[4];
#pragma unroll
  for (int mt = 0; mt < 4; ++mt) {
    const int row = mt * 16 + (lane & 15);
#pragma unroll
    for (int q = 0; q < 8; ++q) {
      const int k0 = ((q < 4) ? 0 : 16) + ((lane >= 16) ? 8 : 0) + (q & 3) * 2;
      a[mt].u[q] = *(const unsigned int*)&aT[row * 40 + k0];
    }
  }
#pragma unroll
  for (int nt = 0; nt < 4; ++nt) {
    const size_t tile = (size_t)(wave * 4 + nt) * nk + kc;
    v16h b = *(const v16h*)(wp + tile * 512 + lane * 16);
#pragma unroll
    for (int mt = 0; mt < 4; ++mt)
      acc[mt][nt] = __builtin_amdgcn_wmma_f32_16x16x32_f16(
          false, a[mt].h, false, b, (short)0, acc[mt][nt], false, false);
  }
}

// One K-chunk of a GEMM whose A panel (64 x 512, stride 520) already sits in LDS.
__device__ __forceinline__ void lds_chunk(const _Float16* __restrict__ mB,
                                          const _Float16* __restrict__ wp,
                                          int nk, int kc, int lane, int wave,
                                          v8f (&acc)[4][4]) {
  AFrag a[4];
#pragma unroll
  for (int mt = 0; mt < 4; ++mt) {
    const int row = mt * 16 + (lane & 15);
#pragma unroll
    for (int q = 0; q < 8; ++q) {
      const int k0 = kc * 32 + ((q < 4) ? 0 : 16) + ((lane >= 16) ? 8 : 0) + (q & 3) * 2;
      a[mt].u[q] = *(const unsigned int*)&mB[row * 520 + k0];
    }
  }
#pragma unroll
  for (int nt = 0; nt < 4; ++nt) {
    const size_t tile = (size_t)(wave * 4 + nt) * nk + kc;
    v16h b = *(const v16h*)(wp + tile * 512 + lane * 16);
#pragma unroll
    for (int mt = 0; mt < 4; ++mt)
      acc[mt][nt] = __builtin_amdgcn_wmma_f32_16x16x32_f16(
          false, a[mt].h, false, b, (short)0, acc[mt][nt], false, false);
  }
}

// ---------------------------------------------------------------------------
// Fused edge-message kernel (~94% of FLOPs).
// Workgroup = 256 threads = 8 waves; tile = 64 edges x 512 outputs.
// Wave w owns output columns [w*64, w*64+64) -> acc[4 m-tiles][4 n-tiles].
// GEMM1 K = 1792 in 3 phases (src / dst / demb) with hoisted base pointers;
// 6 lattice columns folded in by VALU FMA in the epilogue.  GEMM2 reads the
// silu'd panel from LDS (stride 520 halves: conflict-free) and scatter-adds
// f32 atomics into agg[src].
// ---------------------------------------------------------------------------
__global__ __launch_bounds__(256) void edge_kernel(
    const _Float16* __restrict__ hn16, const _Float16* __restrict__ demb16,
    const _Float16* __restrict__ w1p, const _Float16* __restrict__ w2p,
    const float* __restrict__ eW1L,   // this layer's eW1 (for the 6 lat cols)
    const float* __restrict__ b1, const float* __restrict__ b2,
    const int* __restrict__ eidx, const int* __restrict__ n2g,
    const float* __restrict__ lat, float* __restrict__ agg) {
  __shared__ _Float16 aT[EM * 40];     // A staging slab (64 rows x 32 K), padded
  __shared__ _Float16 mB[EM * 520];    // GEMM1 output (64 x 512), padded
  __shared__ float    latS[EM * 6];
  __shared__ int      srcS[EM], dstS[EM];

  const int tid = threadIdx.x, lane = tid & 31, wave = tid >> 5;
  const int e0 = blockIdx.x * EM;

  if (tid < EM) {
    int e = min(e0 + tid, NE - 1);
    srcS[tid] = eidx[e];
    dstS[tid] = eidx[NE + e];
  }
  __syncthreads();
  if (tid < EM) {
    int g = n2g[srcS[tid]];
#pragma unroll
    for (int d = 0; d < 6; ++d) latS[tid * 6 + d] = lat[g * 6 + d];
  }

  // staging geometry: thread handles row r, 8 halves at p*8 (one uint4)
  const int r = tid >> 2, p = tid & 3;
  const int er = min(e0 + r, NE - 1);
  const _Float16* pSrc = hn16 + (size_t)srcS[r] * HD + p * 8;
  const _Float16* pDst = hn16 + (size_t)dstS[r] * HD + p * 8;
  const _Float16* pDmb = demb16 + (size_t)er * DISF + p * 8;
  _Float16* stDst = &aT[r * 40 + p * 8];

  v8f acc[4][4] = {};
  for (int kc = 0; kc < 16; ++kc)
    edge_chunk(pSrc + kc * 32, stDst, aT, w1p, KE_CH, kc, lane, wave, acc);
  for (int kc = 16; kc < 32; ++kc)
    edge_chunk(pDst + (kc - 16) * 32, stDst, aT, w1p, KE_CH, kc, lane, wave, acc);
  for (int kc = 32; kc < KE_CH; ++kc)
    edge_chunk(pDmb + (kc - 32) * 32, stDst, aT, w1p, KE_CH, kc, lane, wave, acc);

  // epilogue 1: lattice columns + bias + silu -> mB  (C layout: vgpr i, lane l
  // -> m = i + 8*(l>=16), n = l%16)
#pragma unroll
  for (int mt = 0; mt < 4; ++mt)
#pragma unroll
    for (int nt = 0; nt < 4; ++nt) {
      const int n = wave * 64 + nt * 16 + (lane & 15);
      const float* wl = eW1L + (size_t)n * DINF + 2 * HD;  // cols 1024..1029
      const float bias = b1[n];
#pragma unroll
      for (int i = 0; i < 8; ++i) {
        const int m = mt * 16 + i + ((lane >= 16) ? 8 : 0);
        float x = acc[mt][nt][i];
#pragma unroll
        for (int d = 0; d < 6; ++d) x += latS[m * 6 + d] * wl[d];
        mB[m * 520 + n] = (_Float16)siluf(x + bias);
      }
    }
  __syncthreads();

  // GEMM2 from LDS
  v8f acc2[4][4] = {};
  for (int kc = 0; kc < KH_CH; ++kc)
    lds_chunk(mB, w2p, KH_CH, kc, lane, wave, acc2);

  // epilogue 2: bias + silu + scatter-add (segment mean numerator)
#pragma unroll
  for (int mt = 0; mt < 4; ++mt)
#pragma unroll
    for (int nt = 0; nt < 4; ++nt) {
      const int n = wave * 64 + nt * 16 + (lane & 15);
      const float bias = b2[n];
#pragma unroll
      for (int i = 0; i < 8; ++i) {
        const int m = mt * 16 + i + ((lane >= 16) ? 8 : 0);
        if (e0 + m < NE) {
          float x = siluf(acc2[mt][nt][i] + bias);
          atomAddF(&agg[(size_t)srcS[m] * HD + n], x);
        }
      }
    }
}

// ---------------------------------------------------------------------------
// Fused node-update kernel: u = silu(silu([hn | agg/cnt] @ nW1.T) @ nW2.T);
// h += u.  32 nodes per workgroup (small share of FLOPs).
// ---------------------------------------------------------------------------
__global__ __launch_bounds__(256) void node_kernel(
    const _Float16* __restrict__ hn16, const float* __restrict__ agg,
    const float* __restrict__ counts, const _Float16* __restrict__ m1p,
    const _Float16* __restrict__ m2p, const float* __restrict__ c1,
    const float* __restrict__ c2, float* __restrict__ h) {
  __shared__ _Float16 aT[32 * 40];
  __shared__ _Float16 mB[32 * 520];

  const int tid = threadIdx.x, lane = tid & 31, wave = tid >> 5;
  const int n0 = blockIdx.x * 32;
  const int r = tid >> 3, p = tid & 7;
  const int nodeR = min(n0 + r, NN - 1);   // clamp tail rows (stores guarded)

  v8f acc[2][4] = {};
  for (int kc = 0; kc < KN_CH; ++kc) {
    __syncthreads();
    const int kb = kc * 32 + p * 4;
    if (kc < 16) {
      *(uint2*)&aT[r * 40 + p * 4] =
          *(const uint2*)(hn16 + (size_t)nodeR * HD + kb);
    } else {
      const float rc = 1.0f / fmaxf(counts[nodeR], 1.0f);
      const float* ap = agg + (size_t)nodeR * HD + (kb - 512);
      _Float16 t4[4];
#pragma unroll
      for (int i = 0; i < 4; ++i) t4[i] = (_Float16)(ap[i] * rc);
      *(uint2*)&aT[r * 40 + p * 4] = *(const uint2*)t4;
    }
    __syncthreads();

    AFrag a[2];
#pragma unroll
    for (int mt = 0; mt < 2; ++mt) {
      const int row = mt * 16 + (lane & 15);
#pragma unroll
      for (int q = 0; q < 8; ++q) {
        const int k0 = ((q < 4) ? 0 : 16) + ((lane >= 16) ? 8 : 0) + (q & 3) * 2;
        a[mt].u[q] = *(const unsigned int*)&aT[row * 40 + k0];
      }
    }
#pragma unroll
    for (int nt = 0; nt < 4; ++nt) {
      const size_t tile = (size_t)(wave * 4 + nt) * KN_CH + kc;
      v16h b = *(const v16h*)(m1p + tile * 512 + lane * 16);
      acc[0][nt] = __builtin_amdgcn_wmma_f32_16x16x32_f16(
          false, a[0].h, false, b, (short)0, acc[0][nt], false, false);
      acc[1][nt] = __builtin_amdgcn_wmma_f32_16x16x32_f16(
          false, a[1].h, false, b, (short)0, acc[1][nt], false, false);
    }
  }

#pragma unroll
  for (int mt = 0; mt < 2; ++mt)
#pragma unroll
    for (int nt = 0; nt < 4; ++nt) {
      const int n = wave * 64 + nt * 16 + (lane & 15);
      const float bias = c1[n];
#pragma unroll
      for (int i = 0; i < 8; ++i) {
        const int m = mt * 16 + i + ((lane >= 16) ? 8 : 0);
        mB[m * 520 + n] = (_Float16)siluf(acc[mt][nt][i] + bias);
      }
    }
  __syncthreads();

  v8f acc2[2][4] = {};
  for (int kc = 0; kc < KH_CH; ++kc) {
    AFrag a[2];
#pragma unroll
    for (int mt = 0; mt < 2; ++mt) {
      const int row = mt * 16 + (lane & 15);
#pragma unroll
      for (int q = 0; q < 8; ++q) {
        const int k0 = kc * 32 + ((q < 4) ? 0 : 16) + ((lane >= 16) ? 8 : 0) + (q & 3) * 2;
        a[mt].u[q] = *(const unsigned int*)&mB[row * 520 + k0];
      }
    }
#pragma unroll
    for (int nt = 0; nt < 4; ++nt) {
      const size_t tile = (size_t)(wave * 4 + nt) * KH_CH + kc;
      v16h b = *(const v16h*)(m2p + tile * 512 + lane * 16);
      acc2[0][nt] = __builtin_amdgcn_wmma_f32_16x16x32_f16(
          false, a[0].h, false, b, (short)0, acc2[0][nt], false, false);
      acc2[1][nt] = __builtin_amdgcn_wmma_f32_16x16x32_f16(
          false, a[1].h, false, b, (short)0, acc2[1][nt], false, false);
    }
  }

#pragma unroll
  for (int mt = 0; mt < 2; ++mt)
#pragma unroll
    for (int nt = 0; nt < 4; ++nt) {
      const int n = wave * 64 + nt * 16 + (lane & 15);
      const float bias = c2[n];
#pragma unroll
      for (int i = 0; i < 8; ++i) {
        const int m = mt * 16 + i + ((lane >= 16) ? 8 : 0);
        const int node = n0 + m;
        if (node < NN) {
          float x = siluf(acc2[mt][nt][i] + bias);
          h[(size_t)node * HD + n] += x;   // residual; unique (node,n) per thread
        }
      }
    }
}

// ---------------------------------------------------------------------------
// Final LN + 3-wide projection: out = LN(h) @ Wc.T   (one wave per node)
// ---------------------------------------------------------------------------
__global__ __launch_bounds__(256) void final_kernel(const float* __restrict__ h,
                                                    const float* __restrict__ g,
                                                    const float* __restrict__ b,
                                                    const float* __restrict__ Wc,
                                                    float* __restrict__ out) {
  int wave = threadIdx.x >> 5, lane = threadIdx.x & 31;
  int node = blockIdx.x * 8 + wave;
  if (node >= NN) return;
  const float* row = h + (size_t)node * HD;
  float x[16], s = 0.f, s2 = 0.f;
#pragma unroll
  for (int i = 0; i < 16; ++i) {
    float v = row[lane + i * 32];
    x[i] = v; s += v; s2 += v * v;
  }
  s = waveRed(s); s2 = waveRed(s2);
  float mu = s * (1.0f / HD);
  float rs = rsqrtf(s2 * (1.0f / HD) - mu * mu + 1e-5f);
  float p0 = 0.f, p1 = 0.f, p2 = 0.f;
#pragma unroll
  for (int i = 0; i < 16; ++i) {
    int c = lane + i * 32;
    float v = (x[i] - mu) * rs * g[c] + b[c];
    p0 += v * Wc[c]; p1 += v * Wc[HD + c]; p2 += v * Wc[2 * HD + c];
  }
  p0 = waveRed(p0); p1 = waveRed(p1); p2 = waveRed(p2);
  if (lane == 0) {
    out[node * 3 + 0] = p0;
    out[node * 3 + 1] = p1;
    out[node * 3 + 2] = p2;
  }
}

// ---------------------------------------------------------------------------
// Host launcher
// ---------------------------------------------------------------------------
extern "C" void kernel_launch(void* const* d_in, const int* in_sizes, int n_in,
                              void* d_out, int out_size, void* d_ws, size_t ws_size,
                              hipStream_t stream) {
  const int*   atom_types = (const int*)d_in[0];
  const int*   node2graph = (const int*)d_in[1];
  const int*   edge_index = (const int*)d_in[2];
  const float* frac_diff  = (const float*)d_in[3];
  const float* lattices   = (const float*)d_in[4];
  const float* t          = (const float*)d_in[5];
  const float* emb_table  = (const float*)d_in[6];
  const float* Wt         = (const float*)d_in[7];
  const float* Wa         = (const float*)d_in[8];
  const float* ba         = (const float*)d_in[9];
  const float* ln_g       = (const float*)d_in[10];
  const float* ln_b       = (const float*)d_in[11];
  const float* eW1        = (const float*)d_in[12];
  const float* eb1        = (const float*)d_in[13];
  const float* eW2        = (const float*)d_in[14];
  const float* eb2        = (const float*)d_in[15];
  const float* nW1        = (const float*)d_in[16];
  const float* nb1        = (const float*)d_in[17];
  const float* nW2        = (const float*)d_in[18];
  const float* nb2        = (const float*)d_in[19];
  const float* fln_g      = (const float*)d_in[20];
  const float* fln_b      = (const float*)d_in[21];
  const float* Wc         = (const float*)d_in[22];
  float* out = (float*)d_out;

  // ---- workspace carve-up (256B aligned) ----
  char* wsb = (char*)d_ws;
  size_t off = 0;
  auto carve = [&](size_t bytes) -> char* {
    char* p = wsb + off;
    off = (off + bytes + 255) & ~(size_t)255;
    return p;
  };
  float*    h      = (float*)   carve((size_t)NN * HD * 4);
  _Float16* hn16   = (_Float16*)carve((size_t)NN * HD * 2);
  float*    agg    = (float*)   carve((size_t)NN * HD * 4);
  float*    counts = (float*)   carve((size_t)NN * 4);
  _Float16* demb16 = (_Float16*)carve((size_t)NE * DISF * 2);
  const size_t W1S = (size_t)32 * KE_CH * 512;   // halves per layer
  const size_t W2S = (size_t)32 * KH_CH * 512;
  const size_t M1S = (size_t)32 * KN_CH * 512;
  const size_t M2S = (size_t)32 * KH_CH * 512;
  _Float16* w1p = (_Float16*)carve(NL * W1S * 2);
  _Float16* w2p = (_Float16*)carve(NL * W2S * 2);
  _Float16* m1p = (_Float16*)carve(NL * M1S * 2);
  _Float16* m2p = (_Float16*)carve(NL * M2S * 2);
  (void)ws_size; (void)n_in; (void)in_sizes; (void)out_size;

  // ---- prep ----
  zero_kernel<<<(NN + 255) / 256, 256, 0, stream>>>(counts, NN);
  counts_kernel<<<(NE + 255) / 256, 256, 0, stream>>>(edge_index, counts);

  {
    int tot = NL * (int)W1S;
    pack_w_kernel<<<(tot + 255) / 256, 256, 0, stream>>>(eW1, w1p, KE_CH, DINF, 1, tot);
    tot = NL * (int)W2S;
    pack_w_kernel<<<(tot + 255) / 256, 256, 0, stream>>>(eW2, w2p, KH_CH, HD, 0, tot);
    tot = NL * (int)M1S;
    pack_w_kernel<<<(tot + 255) / 256, 256, 0, stream>>>(nW1, m1p, KN_CH, 2 * HD, 0, tot);
    tot = NL * (int)M2S;
    pack_w_kernel<<<(tot + 255) / 256, 256, 0, stream>>>(nW2, m2p, KH_CH, HD, 0, tot);
  }

  {
    long long tot = (long long)NE * 384;
    demb_kernel<<<(int)((tot + 255) / 256), 256, 0, stream>>>(frac_diff, demb16);
  }

  init_h_kernel<<<(NN * HD + 255) / 256, 256, 0, stream>>>(
      atom_types, emb_table, Wt, Wa, ba, t, h);

  // ---- layers ----
  for (int layer = 0; layer < NL; ++layer) {
    ln_kernel<<<(NN + 7) / 8, 256, 0, stream>>>(h, ln_g + layer * HD,
                                                ln_b + layer * HD, hn16);
    zero_kernel<<<(NN * HD + 255) / 256, 256, 0, stream>>>(agg, NN * HD);
    edge_kernel<<<(NE + EM - 1) / EM, 256, 0, stream>>>(
        hn16, demb16, w1p + (size_t)layer * W1S, w2p + (size_t)layer * W2S,
        eW1 + (size_t)layer * HD * DINF, eb1 + layer * HD, eb2 + layer * HD,
        edge_index, node2graph, lattices, agg);
    node_kernel<<<(NN + 31) / 32, 256, 0, stream>>>(
        hn16, agg, counts, m1p + (size_t)layer * M1S, m2p + (size_t)layer * M2S,
        nb1 + layer * HD, nb2 + layer * HD, h);
  }

  final_kernel<<<(NN + 7) / 8, 256, 0, stream>>>(h, fln_g, fln_b, Wc, out);
}
